// HamiltonianBlock_85847806313133
// MI455X (gfx1250) — matrix operand order repaired and checked by reading, not compiled, not run on hardware
//
#include <hip/hip_runtime.h>
#include <hip/hip_bf16.h>
#include <math.h>

// ---------------------------------------------------------------------------
// Problem constants (match the reference)
// ---------------------------------------------------------------------------
#define BB   4
#define SS   8192
#define DD   256
#define HID  512
#define RR   32
#define TOK  (BB * SS)          // 32768 tokens
#define DIN  (2 * DD)           // 512
#define H1N  (2 * HID)          // 1024
#define H2N  (HID)              // 512

#define MT   64                 // tokens per workgroup in GEMM kernels
#define NWG  (TOK / MT)         // 512 workgroups

// padded LDS row strides (bf16 elements); keep 16B alignment of rows
#define H1P  (H1N + 8)          // 1032  (2064 B/row, 16B aligned)
#define H2P  (H2N + 8)          // 520   (1040 B/row, 16B aligned)

typedef __attribute__((ext_vector_type(16))) __bf16 v16bf;
typedef __attribute__((ext_vector_type(8)))  float  v8f;

// ---------------------------------------------------------------------------
// bf16 helpers (round-to-nearest-even) — used only OUTSIDE the WMMA hot loops
// ---------------------------------------------------------------------------
__device__ __forceinline__ __bf16 f2bf(float f) {
    union { float f; unsigned u; } in{ f };
    unsigned u = in.u;
    u += 0x7fffu + ((u >> 16) & 1u);
    union { unsigned short s; __bf16 b; } out;
    out.s = (unsigned short)(u >> 16);
    return out.b;
}
__device__ __forceinline__ float bf2f(__bf16 b) {
    union { __bf16 b; unsigned short s; } in{ b };
    union { unsigned u; float f; } out;
    out.u = ((unsigned)in.s) << 16;
    return out.f;
}

// ---------------------------------------------------------------------------
// WMMA fragment loaders — pure 16B vector loads, no conversion VALU.
//
// A (16x32 bf16, M x K): lane m = l&15, hi = l>>4.
//   e in [0,8):  K = kb + 8*hi + e          (8 contiguous bf16 = one b128)
//   e in [8,16): K = kb + 16 + 8*hi + (e-8) (8 contiguous bf16 = one b128)
//
// B (32x16 bf16, K x N): lane n = l&15, hi = l>>4.
//   e in [0,8): K = kb + 8*hi + e; e in [8,16): K = kb + 16 + 8*hi + (e-8) (two b128; [N][K] layout)
//
// C/D (16x16 f32): lane n = l&15, hi = l>>4; vgpr r -> row (r + 8*hi).
// ---------------------------------------------------------------------------
union frag_u { uint4 u[2]; v16bf v; };

__device__ __forceinline__ v16bf load_a_bf16(const __bf16* __restrict__ base,
                                             int ld, int row0, int kb, int lane) {
    int m = lane & 15, hi = lane >> 4;
    const __bf16* r0 = base + (size_t)(row0 + m) * ld + kb + 8 * hi;
    frag_u t;
    t.u[0] = *(const uint4*)(r0);
    t.u[1] = *(const uint4*)(r0 + 16);
    return t.v;
}

__device__ __forceinline__ v16bf load_a_lds(const __bf16* __restrict__ base,
                                            int ld, int row0, int kb, int lane) {
    int m = lane & 15, hi = lane >> 4;
    const __bf16* r0 = base + (size_t)(row0 + m) * ld + kb + 8 * hi;
    frag_u t;
    t.u[0] = *(const uint4*)(r0);
    t.u[1] = *(const uint4*)(r0 + 16);
    return t.v;
}

__device__ __forceinline__ v16bf load_b_bf16(const __bf16* __restrict__ wt,
                                             int ldk, int ncol0, int kb, int lane) {
    int n = lane & 15, hi = lane >> 4;
    const __bf16* r = wt + (size_t)(ncol0 + n) * ldk + kb + 8 * hi;
    frag_u t;
    t.u[0] = *(const uint4*)(r);
    t.u[1] = *(const uint4*)(r + 16);
    return t.v;
}

__device__ __forceinline__ v8f wmma_bf16(v16bf a, v16bf b, v8f c) {
    return __builtin_amdgcn_wmma_f32_16x16x32_bf16(
        false, a, false, b, (short)0, c, false, false);
}

__device__ __forceinline__ float gelu_exact(float x) {
    return 0.5f * x * (1.0f + erff(x * 0.70710678118654752f));
}

__device__ __forceinline__ float block_reduce_256(float v, float* red) {
    int t = threadIdx.x;
    red[t] = v;
    __syncthreads();
#pragma unroll
    for (int s = 128; s > 0; s >>= 1) {
        if (t < s) red[t] += red[t + s];
        __syncthreads();
    }
    float r = red[0];
    __syncthreads();
    return r;
}

// ---------------------------------------------------------------------------
// Prep kernels
// ---------------------------------------------------------------------------
__global__ void zero_acc_kernel(float* acc) {
    if (threadIdx.x < 8) acc[threadIdx.x] = 0.0f;
}

// x = concat(q,p): xb [32768][512] bf16 (cols 0..255 = q, 256..511 = p)
__global__ void conv_x_kernel(const float* __restrict__ q,
                              const float* __restrict__ p,
                              __bf16* __restrict__ xb) {
    const size_t N = (size_t)TOK * DIN;
    for (size_t i = (size_t)blockIdx.x * blockDim.x + threadIdx.x; i < N;
         i += (size_t)gridDim.x * blockDim.x) {
        size_t tok = i >> 9;            // / DIN
        int    c   = (int)(i & (DIN - 1));
        float  v   = (c < DD) ? q[tok * DD + c] : p[tok * DD + (c - DD)];
        xb[i] = f2bf(v);
    }
}

// W1 [512][1024] f32 row-major -> W1t [1024][512] bf16 (N-major, K contiguous)
__global__ void conv_w1_kernel(const float* __restrict__ w, __bf16* __restrict__ wt) {
    for (int i = blockIdx.x * blockDim.x + threadIdx.x; i < DIN * H1N;
         i += gridDim.x * blockDim.x) {
        int k = i / H1N, n = i % H1N;
        wt[(size_t)n * DIN + k] = f2bf(w[i]);
    }
}
// W2 [1024][512] -> W2t [512][1024]
__global__ void conv_w2_kernel(const float* __restrict__ w, __bf16* __restrict__ wt) {
    for (int i = blockIdx.x * blockDim.x + threadIdx.x; i < H1N * H2N;
         i += gridDim.x * blockDim.x) {
        int k = i / H2N, n = i % H2N;
        wt[(size_t)n * H1N + k] = f2bf(w[i]);
    }
}
// low-rank projections: [256][32] f32 -> [32][256] bf16, 4 matrices stacked
// order: 0=ppp_w1, 1=ppp_w2, 2=ppp_w3, 3=qqq_w3
__global__ void conv_wr_kernel(const float* __restrict__ w0, const float* __restrict__ w1,
                               const float* __restrict__ w2, const float* __restrict__ w3,
                               __bf16* __restrict__ wrt) {
    const float* src[4] = { w0, w1, w2, w3 };
    for (int i = blockIdx.x * blockDim.x + threadIdx.x; i < 4 * DD * RR;
         i += gridDim.x * blockDim.x) {
        int m = i / (DD * RR);
        int rem = i - m * (DD * RR);
        int d = rem / RR, r = rem % RR;
        wrt[(size_t)m * (RR * DD) + (size_t)r * DD + d] = f2bf(src[m][rem]);
    }
}

// ---------------------------------------------------------------------------
// q_sum / p_sum in fp32: one block per batch, thread d sums over S (coalesced)
// ---------------------------------------------------------------------------
__global__ __launch_bounds__(DD) void sums_kernel(const float* __restrict__ q,
                                                  const float* __restrict__ p,
                                                  float* __restrict__ qsum,
                                                  float* __restrict__ psum) {
    int b = blockIdx.x, d = threadIdx.x;
    const float* qb = q + (size_t)b * SS * DD + d;
    const float* pb = p + (size_t)b * SS * DD + d;
    float sq = 0.f, sp = 0.f;
    for (int s = 0; s < SS; ++s) {
        sq += qb[(size_t)s * DD];
        sp += pb[(size_t)s * DD];
    }
    qsum[b * DD + d] = sq;
    psum[b * DD + d] = sp;
}

// ---------------------------------------------------------------------------
// Cubic kernel: p1=p@W1, p2=p@W2, p3=p@W3, q3=q@Wq3 for a 64-token tile,
// then accumulate sum(p1*p2*p3) into acc[0] and sum(p1*p2*q3) into acc[1].
// Conversion-free: A from bf16 xb, B from bf16 transposed weights.
// ---------------------------------------------------------------------------
__global__ __launch_bounds__(256) void cubic_kernel(const __bf16* __restrict__ xb,
                                                    const __bf16* __restrict__ wrt,
                                                    float* __restrict__ acc) {
    __shared__ float tiles[4][MT][RR + 1];
    __shared__ float red[256];

    int tok0 = blockIdx.x * MT;
    int wv = threadIdx.x >> 5, lane = threadIdx.x & 31;
    int nlo = lane & 15, hi = lane >> 4;

    for (int t = wv; t < 32; t += 8) {
        int m  = t >> 3;            // which matrix (0..2 -> p, 3 -> q)
        int c  = (t >> 2) & 1;      // column tile (R=32 -> 2 tiles)
        int rt = t & 3;             // row tile (64 tokens -> 4 tiles)
        int koff = (m == 3) ? 0 : DD;   // q = cols [0,256), p = cols [256,512)
        const __bf16* W = wrt + (size_t)m * RR * DD;
        v8f accv = {};
#pragma unroll
        for (int kb = 0; kb < DD; kb += 32) {
            v16bf a  = load_a_bf16(xb, DIN, tok0 + rt * 16, koff + kb, lane);
            v16bf bm = load_b_bf16(W, DD, c * 16, kb, lane);
            accv = wmma_bf16(a, bm, accv);
        }
#pragma unroll
        for (int r = 0; r < 8; ++r)
            tiles[m][rt * 16 + r + 8 * hi][c * 16 + nlo] = accv[r];
    }
    __syncthreads();

    float pp = 0.f, pq = 0.f;
    for (int i = threadIdx.x; i < MT * RR; i += 256) {
        int row = i >> 5, col = i & 31;
        float p12 = tiles[0][row][col] * tiles[1][row][col];
        pp += p12 * tiles[2][row][col];
        pq += p12 * tiles[3][row][col];
    }
    float ppt = block_reduce_256(pp, red);
    float pqt = block_reduce_256(pq, red);
    if (threadIdx.x == 0) {
        atomicAdd(acc + 0, ppt);
        atomicAdd(acc + 1, pqt);
    }
}

// ---------------------------------------------------------------------------
// Fused 3-layer token MLP. 64 tokens per workgroup; h1/h2 staged in LDS as
// bf16; hot loops are pure b128-load + v_wmma. Column tiles are processed in
// pairs (two accumulators) so each A fragment feeds two WMMAs.
// acc[2] accumulates sum over tokens of (h2 @ w3 + b3).
// ---------------------------------------------------------------------------
__global__ __launch_bounds__(256) void mlp_kernel(const __bf16* __restrict__ xb,
                                                  const __bf16* __restrict__ w1t,
                                                  const float* __restrict__ b1,
                                                  const __bf16* __restrict__ w2t,
                                                  const float* __restrict__ b2,
                                                  const float* __restrict__ w3,
                                                  const float* __restrict__ b3,
                                                  float* __restrict__ acc) {
    extern __shared__ char smem[];
    __bf16* h1s = (__bf16*)smem;                                   // [64][H1P]
    __bf16* h2s = (__bf16*)(smem + (size_t)MT * H1P * 2);          // [64][H2P]
    float*  red = (float*)(smem + (size_t)MT * H1P * 2 + (size_t)MT * H2P * 2);

    int tok0 = blockIdx.x * MT;
    int wv = threadIdx.x >> 5, lane = threadIdx.x & 31;
    int nlo = lane & 15, hi = lane >> 4;

    // ---- layer 1: x[64,512] @ W1[512,1024] + b1, GELU -> h1s (bf16) ----
    // wave wv owns columns [wv*128, wv*128+128): 8 tiles, processed in pairs
    for (int cp = 0; cp < 4; ++cp) {
        int ncol0 = wv * 128 + cp * 32;       // tile pair: ncol0, ncol0+16
        for (int rt = 0; rt < 4; ++rt) {
            v8f acc0 = {}, acc1 = {};
#pragma unroll 4
            for (int kb = 0; kb < DIN; kb += 32) {
                v16bf a  = load_a_bf16(xb, DIN, tok0 + rt * 16, kb, lane);
                v16bf b0 = load_b_bf16(w1t, DIN, ncol0,      kb, lane);
                v16bf b1f = load_b_bf16(w1t, DIN, ncol0 + 16, kb, lane);
                acc0 = wmma_bf16(a, b0, acc0);
                acc1 = wmma_bf16(a, b1f, acc1);
            }
            int n0 = ncol0 + nlo, n1 = n0 + 16;
            float bias0 = b1[n0], bias1 = b1[n1];
#pragma unroll
            for (int r = 0; r < 8; ++r) {
                int row = rt * 16 + r + 8 * hi;
                h1s[(size_t)row * H1P + n0] = f2bf(gelu_exact(acc0[r] + bias0));
                h1s[(size_t)row * H1P + n1] = f2bf(gelu_exact(acc1[r] + bias1));
            }
        }
    }
    __syncthreads();

    // ---- layer 2: h1[64,1024] @ W2[1024,512] + b2, GELU -> h2s (bf16) ----
    // wave wv owns columns [wv*64, wv*64+64): 4 tiles, processed in pairs
    for (int cp = 0; cp < 2; ++cp) {
        int ncol0 = wv * 64 + cp * 32;
        for (int rt = 0; rt < 4; ++rt) {
            v8f acc0 = {}, acc1 = {};
#pragma unroll 4
            for (int kb = 0; kb < H1N; kb += 32) {
                v16bf a  = load_a_lds(h1s, H1P, rt * 16, kb, lane);
                v16bf b0 = load_b_bf16(w2t, H1N, ncol0,      kb, lane);
                v16bf b1f = load_b_bf16(w2t, H1N, ncol0 + 16, kb, lane);
                acc0 = wmma_bf16(a, b0, acc0);
                acc1 = wmma_bf16(a, b1f, acc1);
            }
            int n0 = ncol0 + nlo, n1 = n0 + 16;
            float bias0 = b2[n0], bias1 = b2[n1];
#pragma unroll
            for (int r = 0; r < 8; ++r) {
                int row = rt * 16 + r + 8 * hi;
                h2s[(size_t)row * H2P + n0] = f2bf(gelu_exact(acc0[r] + bias0));
                h2s[(size_t)row * H2P + n1] = f2bf(gelu_exact(acc1[r] + bias1));
            }
        }
    }
    __syncthreads();

    // ---- layer 3: per-token dot with w3, sum over tile ----
    int row = threadIdx.x >> 2;        // 64 rows, 4 threads each
    int qtr = threadIdx.x & 3;
    float s = 0.f;
    const __bf16* hr = h2s + (size_t)row * H2P;
    for (int k = qtr * 128; k < qtr * 128 + 128; ++k)
        s += bf2f(hr[k]) * w3[k];
    float tot = block_reduce_256(s, red);
    if (threadIdx.x == 0)
        atomicAdd(acc + 2, tot + (float)MT * b3[0]);
}

// ---------------------------------------------------------------------------
// Final scalar assembly: linear + quadratic terms + accumulated cubic/MLP.
// ---------------------------------------------------------------------------
__global__ __launch_bounds__(DD) void final_kernel(const float* __restrict__ clq,
                                                   const float* __restrict__ clp,
                                                   const float* __restrict__ cqp,
                                                   const float* __restrict__ cqq,
                                                   const float* __restrict__ cpp,
                                                   const float* __restrict__ hoff,
                                                   const float* __restrict__ qsum,
                                                   const float* __restrict__ psum,
                                                   const float* __restrict__ acc,
                                                   float* __restrict__ outH) {
    __shared__ float red[DD];
    int d = threadIdx.x;

    float sqp = 0.f, sqq = 0.f, spp = 0.f;           // C.sum(axis=1)[d]
    const float* rqp = cqp + (size_t)d * DD;
    const float* rqq = cqq + (size_t)d * DD;
    const float* rpp = cpp + (size_t)d * DD;
    for (int k = 0; k < DD; ++k) { sqp += rqp[k]; sqq += rqq[k]; spp += rpp[k]; }

    float part = 0.f;
    for (int b = 0; b < BB; ++b) {
        float qs = qsum[b * DD + d];
        float ps = psum[b * DD + d];
        part += qs * clq[d] + ps * clp[d];
        part += qs * sqp * ps + qs * sqq * qs + ps * spp * ps;
    }
    red[d] = part;
    __syncthreads();
    for (int s = DD / 2; s > 0; s >>= 1) {
        if (d < s) red[d] += red[d + s];
        __syncthreads();
    }
    if (d == 0) {
        float H = (float)BB * hoff[0] + red[0]
                + 3.0f * acc[0]   // cubic ppp
                + acc[1]          // cubic qpq
                + acc[2];         // mlp contribution (incl. b3)
        outH[0] = H;
    }
}

// ---------------------------------------------------------------------------
// Host-side launch
// ---------------------------------------------------------------------------
extern "C" void kernel_launch(void* const* d_in, const int* in_sizes, int n_in,
                              void* d_out, int out_size, void* d_ws, size_t ws_size,
                              hipStream_t stream) {
    (void)in_sizes; (void)n_in; (void)out_size; (void)ws_size;

    const float* q   = (const float*)d_in[0];
    const float* p   = (const float*)d_in[1];
    const float* clq = (const float*)d_in[2];
    const float* clp = (const float*)d_in[3];
    const float* cqp = (const float*)d_in[4];
    const float* cqq = (const float*)d_in[5];
    const float* cpp = (const float*)d_in[6];
    const float* hof = (const float*)d_in[7];
    // d_in[8], d_in[9] (qqq_w1/w2) are dead in the reference (source bug).
    const float* qqq_w3 = (const float*)d_in[10];
    const float* ppp_w1 = (const float*)d_in[11];
    const float* ppp_w2 = (const float*)d_in[12];
    const float* ppp_w3 = (const float*)d_in[13];
    const float* mlp_w1 = (const float*)d_in[14];
    const float* mlp_b1 = (const float*)d_in[15];
    const float* mlp_w2 = (const float*)d_in[16];
    const float* mlp_b2 = (const float*)d_in[17];
    const float* mlp_w3 = (const float*)d_in[18];
    const float* mlp_b3 = (const float*)d_in[19];

    float* out = (float*)d_out;

    // workspace layout
    char* ws = (char*)d_ws;
    float*  acc  = (float*)ws;                       // 8 floats
    float*  qsum = (float*)(ws + 64);                // [4][256]
    float*  psum = (float*)(ws + 64 + 4096);         // [4][256]
    size_t off = 8448;
    __bf16* w1t = (__bf16*)(ws + off); off += (size_t)DIN * H1N * 2;  // [1024][512]
    __bf16* w2t = (__bf16*)(ws + off); off += (size_t)H1N * H2N * 2;  // [512][1024]
    __bf16* wrt = (__bf16*)(ws + off); off += (size_t)4 * RR * DD * 2;// 4x[32][256]
    __bf16* xb  = (__bf16*)(ws + off);                                 // [32768][512]

    // prep
    zero_acc_kernel<<<1, 32, 0, stream>>>(acc);
    conv_x_kernel<<<4096, 256, 0, stream>>>(q, p, xb);
    conv_w1_kernel<<<1024, 256, 0, stream>>>(mlp_w1, w1t);
    conv_w2_kernel<<<1024, 256, 0, stream>>>(mlp_w2, w2t);
    conv_wr_kernel<<<128, 256, 0, stream>>>(ppp_w1, ppp_w2, ppp_w3, qqq_w3, wrt);

    // reductions + heavy compute
    sums_kernel<<<BB, DD, 0, stream>>>(q, p, qsum, psum);
    cubic_kernel<<<NWG, 256, 0, stream>>>(xb, wrt, acc);

    size_t mlp_smem = (size_t)MT * H1P * 2 + (size_t)MT * H2P * 2 + 256 * 4;
    mlp_kernel<<<NWG, 256, mlp_smem, stream>>>(xb, w1t, mlp_b1, w2t, mlp_b2,
                                               mlp_w3, mlp_b3, acc);

    // passthrough outputs: out = [q, p, H]
    const size_t NQP = (size_t)BB * SS * DD;
    hipMemcpyAsync(out,       q, NQP * sizeof(float), hipMemcpyDeviceToDevice, stream);
    hipMemcpyAsync(out + NQP, p, NQP * sizeof(float), hipMemcpyDeviceToDevice, stream);

    // scalar H
    final_kernel<<<1, DD, 0, stream>>>(clq, clp, cqp, cqq, cpp, hof,
                                       qsum, psum, acc, out + 2 * NQP);
}
